// Interact_71880572666182
// MI455X (gfx1250) — matrix operand order repaired and checked by reading, not compiled, not run on hardware
//
#include <hip/hip_runtime.h>
#include <hip/hip_bf16.h>

typedef __bf16 bf16_t;
typedef __attribute__((ext_vector_type(16))) __bf16 v16bf;
typedef __attribute__((ext_vector_type(8)))  float  v8f;

// B=16, T=128 -> 2048 (b,t) blocks; N=64 tokens, D=256, H=512.

__device__ __forceinline__ bf16_t tobf(float x) {
  unsigned u = __builtin_bit_cast(unsigned, x);
  u += 0x7FFFu + ((u >> 16) & 1u);              // round-to-nearest-even
  unsigned short s = (unsigned short)(u >> 16);
  return __builtin_bit_cast(bf16_t, s);
}

union FragU { v16bf f; uint4 q[2]; };
union Pack8 { bf16_t h[8]; uint4 q; };
union Pack4 { bf16_t h[4]; uint2 u; };

// A-fragment (16x32 bf16, M=lane%16): halves 0..15 = K[off..off+15], off=(lane/16)*16
__device__ __forceinline__ v16bf ldsA(const bf16_t* base, int stride, int mt, int kbase, int lane) {
  const bf16_t* p = base + (mt * 16 + (lane & 15)) * stride + kbase + ((lane >> 4) << 4);
  FragU u;
  u.q[0] = *(const uint4*)p;
  u.q[1] = *(const uint4*)(p + 8);
  return u.f;
}

// B-fragment (32x16 bf16) from a matrix stored [ncol][k] row-major (k contiguous per lane)
__device__ __forceinline__ v16bf ldsB(const bf16_t* base, int stride, int nt, int kbase, int lane) {
  const bf16_t* p = base + (nt * 16 + (lane & 15)) * stride + kbase + ((lane >> 4) << 4);
  FragU u;
  u.q[0] = *(const uint4*)p;
  u.q[1] = *(const uint4*)(p + 8);
  return u.f;
}

// B-fragment from fragment-ordered tile buffer (global or LDS): 512 halves/tile, lane-contiguous
__device__ __forceinline__ v16bf gldB(const bf16_t* w, int tile, int lane) {
  const bf16_t* p = w + (size_t)tile * 512 + lane * 16;
  FragU u;
  u.q[0] = ((const uint4*)p)[0];
  u.q[1] = ((const uint4*)p)[1];
  return u.f;
}

__device__ __forceinline__ v8f wmma_bf16(v16bf a, v16bf b, v8f c) {
  return __builtin_amdgcn_wmma_f32_16x16x32_bf16(false, a, false, b, (short)0, c, false, false);
}

// C-tile (lane holds N=lane%16, vgpr j -> M = j + 8*(lane/16)) -> row-major bf16
__device__ __forceinline__ void store_rm(bf16_t* dst, int stride, int mt, int nt, v8f acc, float scale, int lane) {
  int col = nt * 16 + (lane & 15);
  int rb  = mt * 16 + ((lane >> 4) << 3);
#pragma unroll
  for (int j = 0; j < 8; ++j) dst[(rb + j) * stride + col] = tobf(acc[j] * scale);
}

// C-tile -> transposed bf16 [n][m]: one ds_store_b128 per tile per lane
__device__ __forceinline__ void store_tr(bf16_t* dst, int stride, int mt, int nt, v8f acc, int lane) {
  int n  = nt * 16 + (lane & 15);
  int mb = mt * 16 + ((lane >> 4) << 3);
  Pack8 u;
#pragma unroll
  for (int j = 0; j < 8; ++j) u.h[j] = tobf(acc[j]);
  *(uint4*)(dst + n * stride + mb) = u.q;
}

// Async-stage 16 weight tiles (16KB) for H-chunk hc into LDS: tiles (kc*32 + hc*2 + nt),
// kc=0..7, nt=0..1, fragment-ordered.  t in [0,128): each thread moves 8 x 16B.
// Uses GLOBAL_LOAD_ASYNC_TO_LDS_B128 (ASYNCcnt) - no VGPR round trip.
__device__ __forceinline__ void stage_half(bf16_t* dst, const bf16_t* w, int hc, int t) {
  unsigned lds0 = (unsigned)(uintptr_t)dst;
#pragma unroll
  for (int i = 0; i < 8; ++i) {
    int c   = t + i * 128;        // uint4 slot 0..1023
    int tl  = c >> 6;             // local tile 0..15
    int off = c & 63;             // uint4 within tile
    int kc = tl >> 1, nt = tl & 1;
    unsigned voff = (unsigned)(((kc * 32 + hc * 2 + nt) << 10) + off * 16); // bytes
    unsigned ldsa = lds0 + (unsigned)(c * 16);
    asm volatile("global_load_async_to_lds_b128 %0, %1, %2"
                 :: "v"(ldsa), "v"(voff), "s"(w) : "memory");
  }
}

__device__ __forceinline__ void wait_async() {
  asm volatile("s_wait_asynccnt 0" ::: "memory");
}

// Repack weight [K,N] f32 -> bf16 WMMA-B fragment order.
__global__ void pack_weight(const float* __restrict__ W, bf16_t* __restrict__ out, int K, int N) {
  int idx = blockIdx.x * blockDim.x + threadIdx.x;
  if (idx >= K * N) return;
  int tile = idx >> 9;
  int r    = idx & 511;
  int lane = r >> 4;
  int i    = r & 15;
  int ntiles = N >> 4;
  int kc = tile / ntiles, nt = tile % ntiles;
  int k = kc * 32 + ((lane >> 4) << 4) + i;
  int n = nt * 16 + (lane & 15);
  out[idx] = tobf(W[(size_t)k * N + n]);
}

__global__ __launch_bounds__(256)
void fused_interact(const float* __restrict__ x1g, const float* __restrict__ x2g,
                    const float* __restrict__ b1g, const float* __restrict__ b2g,
                    float* __restrict__ outg,
                    const bf16_t* __restrict__ wq1p, const bf16_t* __restrict__ wk1p,
                    const bf16_t* __restrict__ wv1p, const bf16_t* __restrict__ wq2p,
                    const bf16_t* __restrict__ wk2p, const bf16_t* __restrict__ wv2p,
                    const bf16_t* __restrict__ w1p,  const bf16_t* __restrict__ w2p)
{
  extern __shared__ char smem[];
  bf16_t* x1s = (bf16_t*)smem;           // 64x256 (32KB)
  bf16_t* x2s = x1s + 64 * 256;          // 64x256 (32KB)
  bf16_t* hbf = x1s;                     // 64x512 (64KB) aliases x after phase 3
  bf16_t* qs  = x2s + 64 * 256;          // 64x32  (4KB)
  bf16_t* ks  = qs  + 64 * 32;           // 64x32  (4KB)
  bf16_t* stg = ks  + 64 * 32;           // 32KB weight staging (2 x 16 tiles)
  float*  s1  = (float*)stg;             // 64x64 f32 (aliases staging; live only around softmax)
  float*  s2  = s1 + 64 * 64;
  bf16_t* p1  = stg + 16384;             // 64x64 bf16 (8KB)
  bf16_t* p2  = p1 + 64 * 64;            // 8KB
  bf16_t* v1t = p2 + 64 * 64;            // 32x64 (4KB) transposed v chunk
  bf16_t* v2t = v1t + 32 * 64;           // 4KB
  bf16_t* as1 = v2t + 32 * 64;           // 64x32 (4KB)
  bf16_t* as2 = as1 + 64 * 32;           // 4KB
  // total 136KB

  const int tid  = threadIdx.x;
  const int lane = tid & 31;
  const int w    = tid >> 5;
  const int bt   = blockIdx.x;
  const int pmt  = w >> 1, pnt = w & 1;          // projection tile (4x2 grid of 16x16)
  const int smt  = w >> 1, snt0 = (w & 1) * 2;   // score tiles (4x4 grid, 2 per wave)

  // ---- Phase 0: x1,x2 -> LDS bf16 ----
  {
    const float4* g1 = (const float4*)(x1g + (size_t)bt * 64 * 256);
    const float4* g2 = (const float4*)(x2g + (size_t)bt * 64 * 256);
#pragma unroll
    for (int i = 0; i < 16; ++i) {
      int idx = tid + i * 256;
      float4 a = g1[idx];
      Pack4 pa; pa.h[0] = tobf(a.x); pa.h[1] = tobf(a.y); pa.h[2] = tobf(a.z); pa.h[3] = tobf(a.w);
      *(uint2*)(x1s + idx * 4) = pa.u;
      float4 b = g2[idx];
      Pack4 pb; pb.h[0] = tobf(b.x); pb.h[1] = tobf(b.y); pb.h[2] = tobf(b.z); pb.h[3] = tobf(b.w);
      *(uint2*)(x2s + idx * 4) = pb.u;
    }
  }
  __syncthreads();

  // ---- Phase 1: scores (kept in registers; staging buffer aliases score memory) ----
  const float scale = 0.0625f; // 256^-0.5
  v8f sAll[4] = {{}, {}, {}, {}};   // [0..1]=s1 tiles, [2..3]=s2 tiles
#pragma unroll
  for (int sidx = 0; sidx < 2; ++sidx) {
    const bf16_t* xq = sidx ? x2s : x1s;
    const bf16_t* xk = sidx ? x1s : x2s;
    const bf16_t* Wq = sidx ? wq2p : wq1p;
    const bf16_t* Wk = sidx ? wk1p : wk2p;   // s1 pairs q1 with k2; s2 pairs q2 with k1
#pragma unroll 1
    for (int hc = 0; hc < 16; ++hc) {
      // stage this chunk's q & k weight tiles into LDS (async, once per block)
      if (tid < 128) stage_half(stg, Wq, hc, tid);
      else           stage_half(stg + 8192, Wk, hc, tid - 128);
      wait_async();
      __syncthreads();

      v8f aq = {}; v8f ak = {};
#pragma unroll
      for (int kc = 0; kc < 8; ++kc) {
        int tl = kc * 2 + pnt;
        aq = wmma_bf16(ldsA(xq, 256, pmt, kc * 32, lane), gldB(stg,        tl, lane), aq);
        ak = wmma_bf16(ldsA(xk, 256, pmt, kc * 32, lane), gldB(stg + 8192, tl, lane), ak);
      }
      store_rm(qs, 32, pmt, pnt, aq, scale, lane);
      store_rm(ks, 32, pmt, pnt, ak, 1.0f, lane);
      __syncthreads();
      {
        v16bf A = ldsA(qs, 32, smt, 0, lane);
        sAll[sidx * 2 + 0] = wmma_bf16(A, ldsB(ks, 32, snt0,     0, lane), sAll[sidx * 2 + 0]);
        sAll[sidx * 2 + 1] = wmma_bf16(A, ldsB(ks, 32, snt0 + 1, 0, lane), sAll[sidx * 2 + 1]);
      }
      __syncthreads();
    }
  }
  // staging region now dead: write scores f32
  {
    int col = snt0 * 16 + (lane & 15);
    int rb  = smt * 16 + ((lane >> 4) << 3);
#pragma unroll
    for (int j = 0; j < 8; ++j) {
      s1[(rb + j) * 64 + col]      = sAll[0][j];
      s1[(rb + j) * 64 + col + 16] = sAll[1][j];
      s2[(rb + j) * 64 + col]      = sAll[2][j];
      s2[(rb + j) * 64 + col + 16] = sAll[3][j];
    }
  }
  __syncthreads();

  // ---- Phase 2: row softmax -> bf16 prob matrices ----
  if (tid < 128) {
    const float* srow = (tid < 64) ? (s1 + tid * 64) : (s2 + (tid - 64) * 64);
    bf16_t*      prow = (tid < 64) ? (p1 + tid * 64) : (p2 + (tid - 64) * 64);
    float mx = -3.4e38f;
    for (int m = 0; m < 64; ++m) mx = fmaxf(mx, srow[m]);
    float sum = 0.f;
    for (int m = 0; m < 64; ++m) sum += __expf(srow[m] - mx);
    float inv = 1.f / sum;
    for (int m = 0; m < 64; ++m) prow[m] = tobf(__expf(srow[m] - mx) * inv);
  }
  __syncthreads();

  // ---- Phase 3: stream v chunks, aggregate, accumulate MLP hidden in registers ----
  v8f hacc[16]; // wave owns n16 in [w*4, w*4+4), all 4 mtiles
#pragma unroll
  for (int i = 0; i < 4; ++i) {
    float bv = b1g[(w * 4 + i) * 16 + (lane & 15)];
#pragma unroll
    for (int mt = 0; mt < 4; ++mt)
#pragma unroll
      for (int j = 0; j < 8; ++j) hacc[mt * 4 + i][j] = bv;
  }
#pragma unroll 1
  for (int hc = 0; hc < 16; ++hc) {
    // stage v weight tiles (scores dead -> staging region reusable)
    if (tid < 128) stage_half(stg, wv1p, hc, tid);
    else           stage_half(stg + 8192, wv2p, hc, tid - 128);
    wait_async();
    __syncthreads();

    v8f av1 = {}; v8f av2 = {};
#pragma unroll
    for (int kc = 0; kc < 8; ++kc) {
      int tl = kc * 2 + pnt;
      av1 = wmma_bf16(ldsA(x1s, 256, pmt, kc * 32, lane), gldB(stg,        tl, lane), av1);
      av2 = wmma_bf16(ldsA(x2s, 256, pmt, kc * 32, lane), gldB(stg + 8192, tl, lane), av2);
    }
    store_tr(v1t, 64, pmt, pnt, av1, lane);
    store_tr(v2t, 64, pmt, pnt, av2, lane);
    __syncthreads();

    v8f aa1 = {}; v8f aa2 = {};
#pragma unroll
    for (int kc = 0; kc < 2; ++kc) {
      aa1 = wmma_bf16(ldsA(p1, 64, pmt, kc * 32, lane), ldsB(v1t, 64, pnt, kc * 32, lane), aa1);
      aa2 = wmma_bf16(ldsA(p2, 64, pmt, kc * 32, lane), ldsB(v2t, 64, pnt, kc * 32, lane), aa2);
    }
    store_rm(as1, 32, pmt, pnt, aa1, 1.0f, lane);
    store_rm(as2, 32, pmt, pnt, aa2, 1.0f, lane);
    __syncthreads();

    // MLP accumulate; preload B-fragments once and sweep all 4 A-tiles (4x L2 reuse)
    {
      v16bf Bf[4];
#pragma unroll
      for (int i = 0; i < 4; ++i) Bf[i] = gldB(w1p, hc * 32 + w * 4 + i, lane);
#pragma unroll
      for (int mt = 0; mt < 4; ++mt) {
        v16bf A1 = ldsA(as1, 32, mt, 0, lane);
#pragma unroll
        for (int i = 0; i < 4; ++i) hacc[mt * 4 + i] = wmma_bf16(A1, Bf[i], hacc[mt * 4 + i]);
      }
#pragma unroll
      for (int i = 0; i < 4; ++i) Bf[i] = gldB(w1p, (16 + hc) * 32 + w * 4 + i, lane);
#pragma unroll
      for (int mt = 0; mt < 4; ++mt) {
        v16bf A2 = ldsA(as2, 32, mt, 0, lane);
#pragma unroll
        for (int i = 0; i < 4; ++i) hacc[mt * 4 + i] = wmma_bf16(A2, Bf[i], hacc[mt * 4 + i]);
      }
    }
    __syncthreads();
  }

  // ---- ReLU -> bf16 LDS (aliases dead x buffers) ----
#pragma unroll
  for (int mt = 0; mt < 4; ++mt)
#pragma unroll
    for (int i = 0; i < 4; ++i) {
      int col = (w * 4 + i) * 16 + (lane & 15);
      int rb  = mt * 16 + ((lane >> 4) << 3);
#pragma unroll
      for (int j = 0; j < 8; ++j)
        hbf[(rb + j) * 512 + col] = tobf(fmaxf(hacc[mt * 4 + i][j], 0.f));
    }
  __syncthreads();

  // ---- Phase 4: out = relu(h) @ W2 + b2 (B-fragments preloaded per kc, 4x reuse) ----
  {
    v8f oacc[16];
#pragma unroll
    for (int i = 0; i < 4; ++i) {
      float bv = b2g[(w * 4 + i) * 16 + (lane & 15)];
#pragma unroll
      for (int mt = 0; mt < 4; ++mt)
#pragma unroll
        for (int j = 0; j < 8; ++j) oacc[mt * 4 + i][j] = bv;
    }
#pragma unroll 1
    for (int kc = 0; kc < 16; ++kc) {
      v16bf Bf[4];
#pragma unroll
      for (int i = 0; i < 4; ++i) Bf[i] = gldB(w2p, kc * 32 + w * 4 + i, lane);
#pragma unroll
      for (int mt = 0; mt < 4; ++mt) {
        v16bf A = ldsA(hbf, 512, mt, kc * 32, lane);
#pragma unroll
        for (int i = 0; i < 4; ++i) oacc[mt * 4 + i] = wmma_bf16(A, Bf[i], oacc[mt * 4 + i]);
      }
    }
    float* outp = outg + (size_t)bt * 64 * 512;
#pragma unroll
    for (int mt = 0; mt < 4; ++mt)
#pragma unroll
      for (int i = 0; i < 4; ++i) {
        int col = (w * 4 + i) * 16 + (lane & 15);
        int rb  = mt * 16 + ((lane >> 4) << 3);
#pragma unroll
        for (int j = 0; j < 8; ++j)
          outp[(size_t)(rb + j) * 512 + col] = oacc[mt * 4 + i][j];
      }
  }
}

extern "C" void kernel_launch(void* const* d_in, const int* in_sizes, int n_in,
                              void* d_out, int out_size, void* d_ws, size_t ws_size,
                              hipStream_t stream) {
  (void)in_sizes; (void)n_in; (void)out_size; (void)ws_size;
  const float* x1  = (const float*)d_in[0];
  const float* x2  = (const float*)d_in[1];
  const float* Wq1 = (const float*)d_in[2];
  const float* Wk1 = (const float*)d_in[3];
  const float* Wv1 = (const float*)d_in[4];
  const float* Wq2 = (const float*)d_in[5];
  const float* Wk2 = (const float*)d_in[6];
  const float* Wv2 = (const float*)d_in[7];
  const float* W1  = (const float*)d_in[8];
  const float* b1  = (const float*)d_in[9];
  const float* W2  = (const float*)d_in[10];
  const float* b2  = (const float*)d_in[11];
  float* out = (float*)d_out;

  bf16_t* ws = (bf16_t*)d_ws;
  const size_t QKV = 256 * 512;
  bf16_t* wq1p = ws;
  bf16_t* wk1p = wq1p + QKV;
  bf16_t* wv1p = wk1p + QKV;
  bf16_t* wq2p = wv1p + QKV;
  bf16_t* wk2p = wq2p + QKV;
  bf16_t* wv2p = wk2p + QKV;
  bf16_t* w1p  = wv2p + QKV;            // 1024x512
  bf16_t* w2p  = w1p + 1024 * 512;      // 512x512

  pack_weight<<<512, 256, 0, stream>>>(Wq1, wq1p, 256, 512);
  pack_weight<<<512, 256, 0, stream>>>(Wk1, wk1p, 256, 512);
  pack_weight<<<512, 256, 0, stream>>>(Wv1, wv1p, 256, 512);
  pack_weight<<<512, 256, 0, stream>>>(Wq2, wq2p, 256, 512);
  pack_weight<<<512, 256, 0, stream>>>(Wk2, wk2p, 256, 512);
  pack_weight<<<512, 256, 0, stream>>>(Wv2, wv2p, 256, 512);
  pack_weight<<<2048, 256, 0, stream>>>(W1, w1p, 1024, 512);
  pack_weight<<<1024, 256, 0, stream>>>(W2, w2p, 512, 512);

  const int SMEM = 139264; // 136 KB dynamic LDS
  hipFuncSetAttribute((const void*)&fused_interact,
                      hipFuncAttributeMaxDynamicSharedMemorySize, SMEM);
  fused_interact<<<2048, 256, SMEM, stream>>>(x1, x2, b1, b2, out,
                                              wq1p, wk1p, wv1p, wq2p, wk2p, wv2p, w1p, w2p);
}
